// BaseAttention_49417893707930
// MI455X (gfx1250) — matrix operand order repaired and checked by reading, not compiled, not run on hardware
//
#include <hip/hip_runtime.h>
#include <math.h>

// ---------------------------------------------------------------------------
// Linear-attention block for MI455X (gfx1250, wave32, WMMA).
// All heavy GEMMs use v_wmma_f32_16x16x32_bf16 (fp32 accumulate).
// Weight-pair fusion: x@Wq@Wq2 == x@(Wq@Wq2), saving 3 of the 7 big GEMMs.
// Inner loops are software-pipelined: fragments for K-step kk+32 are loaded
// into a second register set before the WMMAs for step kk issue, so the
// s_wait before each WMMA block only covers already-aged loads.
// ---------------------------------------------------------------------------

typedef __attribute__((ext_vector_type(16))) __bf16        v16bf;
typedef __attribute__((ext_vector_type(8)))  float         v8f;
typedef __attribute__((ext_vector_type(4)))  unsigned int  u32x4;

union FragU { u32x4 u[2]; v16bf v; };

__device__ __forceinline__ unsigned short f2bf(float f) {
  unsigned int u = __float_as_uint(f);
  u += 0x7FFFu + ((u >> 16) & 1u);          // round-to-nearest-even
  return (unsigned short)(u >> 16);
}
__device__ __forceinline__ float bf2f(unsigned short b) {
  return __uint_as_float(((unsigned int)b) << 16);
}

// Lane base pointer for an A fragment (16x32 bf16, row-major source):
// lane m (0..15) holds row m0+m, K = {kk..kk+7, kk+16..kk+23};
// lanes 16..31 hold the same rows with K offset +8 (per ISA layout table).
__device__ __forceinline__ const unsigned short*
laneA(const unsigned short* A, int lda, int m0, int lane) {
  return A + (size_t)(m0 + (lane & 15)) * lda + (lane >> 4) * 8;
}
__device__ __forceinline__ v16bf loadA(const unsigned short* p, int kk) {
  FragU f;
  f.u[0] = *(const u32x4*)(p + kk);
  f.u[1] = *(const u32x4*)(p + kk + 16);
  return f.v;
}

// Lane base pointer for a B fragment (32x16 bf16, source stored [N][K]):
// lane n (0..15) holds column n0+n, K = {kk..kk+7, kk+16..kk+23}; lanes 16..31: +8.
__device__ __forceinline__ const unsigned short*
laneB(const unsigned short* Bt, int ldb, int n0, int lane) {
  return Bt + (size_t)(n0 + (lane & 15)) * ldb + (lane >> 4) * 8;
}
__device__ __forceinline__ v16bf loadB(const unsigned short* p, int kk) {
  FragU f;
  f.u[0] = *(const u32x4*)(p + kk);
  f.u[1] = *(const u32x4*)(p + kk + 16);
  return f.v;
}

__device__ __forceinline__ v8f wmma_bf16(v16bf a, v16bf b, v8f c) {
  // (neg_a, A, neg_b, B, c_mod, C, reuse_a, reuse_b)
  return __builtin_amdgcn_wmma_f32_16x16x32_bf16(false, a, false, b,
                                                 (short)0, c, false, false);
}

// ---------------------------------------------------------------------------
// Converters / small prep kernels
// ---------------------------------------------------------------------------
__global__ void k_cvt_bf16(const float* __restrict__ in,
                           unsigned short* __restrict__ out, int n) {
  int i = blockIdx.x * 256 + threadIdx.x;
  if (i < n) out[i] = f2bf(in[i]);
}

// in: fp32 [1024][1024] (row = k, col = n) -> out: bf16 [n][k]
__global__ void k_transpose_cvt(const float* __restrict__ in,
                                unsigned short* __restrict__ out) {
  int idx = blockIdx.x * 256 + threadIdx.x;       // 0..1M
  int k = idx >> 10, n = idx & 1023;
  out[(size_t)n * 1024 + k] = f2bf(in[idx]);
}

// bfused[n] = sum_k b1[k] * W2[k][n] + b2[n]
__global__ void k_bias_fuse(const float* __restrict__ b1,
                            const float* __restrict__ W2,
                            const float* __restrict__ b2,
                            float* __restrict__ outb) {
  int n = blockIdx.x * 256 + threadIdx.x;         // grid 4 -> 1024
  float s = b2[n];
  for (int k = 0; k < 1024; ++k) s = fmaf(b1[k], W2[(size_t)k * 1024 + n], s);
  outb[n] = s;
}

// ---------------------------------------------------------------------------
// Shared pipelined 64x32-per-wave GEMM body over K=1024.
// pA[4]: lane pointers for 4 A row-tiles; pB[2]: lane pointers for 2 B
// column-tiles; acc[4][2] accumulators. Double-buffered fragments.
// ---------------------------------------------------------------------------
__device__ __forceinline__ void gemm_body_1024(const unsigned short* pA0,
                                               const unsigned short* pA1,
                                               const unsigned short* pA2,
                                               const unsigned short* pA3,
                                               const unsigned short* pB0,
                                               const unsigned short* pB1,
                                               v8f acc[4][2]) {
  v16bf aC0 = loadA(pA0, 0), aC1 = loadA(pA1, 0);
  v16bf aC2 = loadA(pA2, 0), aC3 = loadA(pA3, 0);
  v16bf bC0 = loadB(pB0, 0), bC1 = loadB(pB1, 0);
  for (int kk = 0; kk < 1024 - 32; kk += 32) {
    const int kn = kk + 32;
    // prefetch far-ahead cachelines of the (shared) B operand
    if (kn + 256 < 1024) {
      __builtin_prefetch(pB0 + kn + 256, 0, 3);
      __builtin_prefetch(pB1 + kn + 256, 0, 3);
    }
    // issue next K-step loads before consuming current fragments
    v16bf aN0 = loadA(pA0, kn), aN1 = loadA(pA1, kn);
    v16bf aN2 = loadA(pA2, kn), aN3 = loadA(pA3, kn);
    v16bf bN0 = loadB(pB0, kn), bN1 = loadB(pB1, kn);
    acc[0][0] = wmma_bf16(aC0, bC0, acc[0][0]);
    acc[0][1] = wmma_bf16(aC0, bC1, acc[0][1]);
    acc[1][0] = wmma_bf16(aC1, bC0, acc[1][0]);
    acc[1][1] = wmma_bf16(aC1, bC1, acc[1][1]);
    acc[2][0] = wmma_bf16(aC2, bC0, acc[2][0]);
    acc[2][1] = wmma_bf16(aC2, bC1, acc[2][1]);
    acc[3][0] = wmma_bf16(aC3, bC0, acc[3][0]);
    acc[3][1] = wmma_bf16(aC3, bC1, acc[3][1]);
    aC0 = aN0; aC1 = aN1; aC2 = aN2; aC3 = aN3;
    bC0 = bN0; bC1 = bN1;
  }
  acc[0][0] = wmma_bf16(aC0, bC0, acc[0][0]);
  acc[0][1] = wmma_bf16(aC0, bC1, acc[0][1]);
  acc[1][0] = wmma_bf16(aC1, bC0, acc[1][0]);
  acc[1][1] = wmma_bf16(aC1, bC1, acc[1][1]);
  acc[2][0] = wmma_bf16(aC2, bC0, acc[2][0]);
  acc[2][1] = wmma_bf16(aC2, bC1, acc[2][1]);
  acc[3][0] = wmma_bf16(aC3, bC0, acc[3][0]);
  acc[3][1] = wmma_bf16(aC3, bC1, acc[3][1]);
}

// ---------------------------------------------------------------------------
// Weight fusion GEMM: Wf[k][n] = sum_j Wa[k][j] * W2[j][n], stored transposed
// as WfT[n][k] bf16 (B-operand layout for the main GEMM). M=N=K=1024, 3 sets.
// ---------------------------------------------------------------------------
__global__ __launch_bounds__(256)
void k_fuse_weights(const unsigned short* __restrict__ Wa,
                    const unsigned short* __restrict__ W2T,
                    unsigned short* __restrict__ WfT) {
  const int z = blockIdx.z;
  const unsigned short* A  = Wa  + (size_t)z * 1024 * 1024;
  const unsigned short* Bt = W2T + (size_t)z * 1024 * 1024;
  unsigned short* out      = WfT + (size_t)z * 1024 * 1024;
  const int lane = threadIdx.x & 31, wave = threadIdx.x >> 5;
  const int m0 = blockIdx.y * 128 + (wave >> 2) * 64;
  const int n0 = blockIdx.x * 128 + (wave & 3) * 32;
  v8f acc[4][2] = {};
  gemm_body_1024(laneA(A, 1024, m0, lane),      laneA(A, 1024, m0 + 16, lane),
                 laneA(A, 1024, m0 + 32, lane), laneA(A, 1024, m0 + 48, lane),
                 laneB(Bt, 1024, n0, lane),     laneB(Bt, 1024, n0 + 16, lane),
                 acc);
#pragma unroll
  for (int i = 0; i < 4; ++i)
#pragma unroll
    for (int j = 0; j < 2; ++j) {
      const int col = n0 + j * 16 + (lane & 15);
#pragma unroll
      for (int r = 0; r < 8; ++r) {
        const int row = m0 + i * 16 + (lane >> 4) * 8 + r;
        out[(size_t)col * 1024 + row] = f2bf(acc[i][j][r]);   // [n][k]
      }
    }
}

// ---------------------------------------------------------------------------
// Main fused QKV GEMM: [16384 x 1024] x [1024 x 3072] (+bias, elu+1 on q,k).
// Epilogue scatters per-head: Qt/Kt as [bh][d][l] (pre-transposed for the
// state GEMM), Vh as [bh][l][d].
// ---------------------------------------------------------------------------
__global__ __launch_bounds__(256)
void k_gemm_qkv(const unsigned short* __restrict__ xbf,
                const unsigned short* __restrict__ WfT,
                const float* __restrict__ bfused,
                unsigned short* __restrict__ Qt,
                unsigned short* __restrict__ Kt,
                unsigned short* __restrict__ Vh) {
  const int lane = threadIdx.x & 31, wave = threadIdx.x >> 5;
  const int m0 = blockIdx.y * 128 + (wave >> 2) * 64;
  const int n0 = blockIdx.x * 128 + (wave & 3) * 32;
  v8f acc[4][2] = {};
  gemm_body_1024(laneA(xbf, 1024, m0, lane),      laneA(xbf, 1024, m0 + 16, lane),
                 laneA(xbf, 1024, m0 + 32, lane), laneA(xbf, 1024, m0 + 48, lane),
                 laneB(WfT, 1024, n0, lane),      laneB(WfT, 1024, n0 + 16, lane),
                 acc);
#pragma unroll
  for (int i = 0; i < 4; ++i)
#pragma unroll
    for (int j = 0; j < 2; ++j) {
      const int col  = n0 + j * 16 + (lane & 15);   // 0..3071
      const int sect = col >> 10;                   // 0=q, 1=k, 2=v
      const int nc   = col & 1023;
      const int h = nc >> 6, d = nc & 63;
      const float bias = bfused[col];
#pragma unroll
      for (int r = 0; r < 8; ++r) {
        const int m = m0 + i * 16 + (lane >> 4) * 8 + r;
        const int bb = m >> 12, l = m & 4095;
        float v = acc[i][j][r] + bias;
        if (sect < 2) v = (v > 0.0f) ? (v + 1.0f) : __expf(v);  // elu(v)+1
        const unsigned short bv = f2bf(v);
        const int bh = bb * 16 + h;
        if (sect == 0)      Qt[((size_t)bh * 64 + d) * 4096 + l] = bv;
        else if (sect == 1) Kt[((size_t)bh * 64 + d) * 4096 + l] = bv;
        else                Vh[((size_t)bh * 4096 + l) * 64 + d] = bv;
      }
    }
}

// ---------------------------------------------------------------------------
// Per-head state: St[bh][e][d] = sum_l q[l][e] * k[l][d]
// A = Qt[bh] (64 x 4096 row-major), B source = Kt[bh] ([d][l] == [N][K]).
// One block (4 waves) per (b,h); wave owns one 16-row e-tile x 4 d-tiles.
// ---------------------------------------------------------------------------
__global__ __launch_bounds__(128)
void k_attn_state(const unsigned short* __restrict__ Qt,
                  const unsigned short* __restrict__ Kt,
                  unsigned short* __restrict__ St) {
  const int bh = blockIdx.x;
  const unsigned short* A  = Qt + (size_t)bh * 64 * 4096;
  const unsigned short* Bt = Kt + (size_t)bh * 64 * 4096;
  const int lane = threadIdx.x & 31, wave = threadIdx.x >> 5;
  const int e0 = wave * 16;
  const unsigned short* pA  = laneA(A, 4096, e0, lane);
  const unsigned short* pB0 = laneB(Bt, 4096, 0,  lane);
  const unsigned short* pB1 = laneB(Bt, 4096, 16, lane);
  const unsigned short* pB2 = laneB(Bt, 4096, 32, lane);
  const unsigned short* pB3 = laneB(Bt, 4096, 48, lane);
  v8f acc[4] = {};
  v16bf aC  = loadA(pA, 0);
  v16bf bC0 = loadB(pB0, 0), bC1 = loadB(pB1, 0);
  v16bf bC2 = loadB(pB2, 0), bC3 = loadB(pB3, 0);
  for (int kk = 0; kk < 4096 - 32; kk += 32) {
    const int kn = kk + 32;
    v16bf aN  = loadA(pA, kn);
    v16bf bN0 = loadB(pB0, kn), bN1 = loadB(pB1, kn);
    v16bf bN2 = loadB(pB2, kn), bN3 = loadB(pB3, kn);
    acc[0] = wmma_bf16(aC, bC0, acc[0]);
    acc[1] = wmma_bf16(aC, bC1, acc[1]);
    acc[2] = wmma_bf16(aC, bC2, acc[2]);
    acc[3] = wmma_bf16(aC, bC3, acc[3]);
    aC = aN; bC0 = bN0; bC1 = bN1; bC2 = bN2; bC3 = bN3;
  }
  acc[0] = wmma_bf16(aC, bC0, acc[0]);
  acc[1] = wmma_bf16(aC, bC1, acc[1]);
  acc[2] = wmma_bf16(aC, bC2, acc[2]);
  acc[3] = wmma_bf16(aC, bC3, acc[3]);
#pragma unroll
  for (int j = 0; j < 4; ++j)
#pragma unroll
    for (int r = 0; r < 8; ++r) {
      const int e = e0 + (lane >> 4) * 8 + r;
      const int d = j * 16 + (lane & 15);
      St[(size_t)bh * 4096 + e * 64 + d] = f2bf(acc[j][r]);   // [e][d]
    }
}

// qks[bh][l] = sum_d q[l][d]*k[l][d] + 1e-6 (coalesced over l per d-row)
__global__ void k_qks(const unsigned short* __restrict__ Qt,
                      const unsigned short* __restrict__ Kt,
                      float* __restrict__ qks) {
  const int bh = blockIdx.y;
  const int l  = blockIdx.x * 256 + threadIdx.x;
  const unsigned short* q = Qt + (size_t)bh * 64 * 4096 + l;
  const unsigned short* k = Kt + (size_t)bh * 64 * 4096 + l;
  float s = 1e-6f;
  for (int d = 0; d < 64; ++d)
    s = fmaf(bf2f(q[(size_t)d * 4096]), bf2f(k[(size_t)d * 4096]), s);
  qks[(size_t)bh * 4096 + l] = s;
}

// ---------------------------------------------------------------------------
// out[l][e] = (sum_d Vh[l][d] * S[d][e]) / qks[l]; scatter to Comb[b][l][h*64+e]
// A = Vh[bh] (4096 x 64), B source = St[bh] ([e][d] == [N][K]). K = 64:
// both K-steps' fragments are loaded up front (no pipelining needed).
// ---------------------------------------------------------------------------
__global__ __launch_bounds__(256)
void k_attn_out(const unsigned short* __restrict__ Vh,
                const unsigned short* __restrict__ St,
                const float* __restrict__ qks,
                unsigned short* __restrict__ Comb) {
  const int bh = blockIdx.y;
  const int bb = bh >> 4, h = bh & 15;
  const unsigned short* A  = Vh + (size_t)bh * 4096 * 64;
  const unsigned short* Bt = St + (size_t)bh * 4096;
  const int lane = threadIdx.x & 31, wave = threadIdx.x >> 5;
  const int m0 = blockIdx.x * 128 + wave * 16;
  const unsigned short* pA = laneA(A, 64, m0, lane);
  v16bf a0 = loadA(pA, 0), a1 = loadA(pA, 32);
  v16bf b00 = loadB(laneB(Bt, 64, 0,  lane), 0);
  v16bf b01 = loadB(laneB(Bt, 64, 16, lane), 0);
  v16bf b02 = loadB(laneB(Bt, 64, 32, lane), 0);
  v16bf b03 = loadB(laneB(Bt, 64, 48, lane), 0);
  v16bf b10 = loadB(laneB(Bt, 64, 0,  lane), 32);
  v16bf b11 = loadB(laneB(Bt, 64, 16, lane), 32);
  v16bf b12 = loadB(laneB(Bt, 64, 32, lane), 32);
  v16bf b13 = loadB(laneB(Bt, 64, 48, lane), 32);
  v8f acc[4] = {};
  acc[0] = wmma_bf16(a0, b00, acc[0]);
  acc[1] = wmma_bf16(a0, b01, acc[1]);
  acc[2] = wmma_bf16(a0, b02, acc[2]);
  acc[3] = wmma_bf16(a0, b03, acc[3]);
  acc[0] = wmma_bf16(a1, b10, acc[0]);
  acc[1] = wmma_bf16(a1, b11, acc[1]);
  acc[2] = wmma_bf16(a1, b12, acc[2]);
  acc[3] = wmma_bf16(a1, b13, acc[3]);
#pragma unroll
  for (int j = 0; j < 4; ++j)
#pragma unroll
    for (int r = 0; r < 8; ++r) {
      const int l = m0 + (lane >> 4) * 8 + r;
      const int e = j * 16 + (lane & 15);
      const float v = acc[j][r] / qks[(size_t)bh * 4096 + l];
      Comb[((size_t)(bb * 4096 + l)) * 1024 + h * 64 + e] = f2bf(v);
    }
}

// ---------------------------------------------------------------------------
// Output GEMM: R[m][n] = Comb @ Wo + bo + x (residual), fp32 out.
// ---------------------------------------------------------------------------
__global__ __launch_bounds__(256)
void k_gemm_out(const unsigned short* __restrict__ Comb,
                const unsigned short* __restrict__ WoT,
                const float* __restrict__ bo,
                const float* __restrict__ x,
                float* __restrict__ R) {
  const int lane = threadIdx.x & 31, wave = threadIdx.x >> 5;
  const int m0 = blockIdx.y * 128 + (wave >> 2) * 64;
  const int n0 = blockIdx.x * 128 + (wave & 3) * 32;
  v8f acc[4][2] = {};
  gemm_body_1024(laneA(Comb, 1024, m0, lane),      laneA(Comb, 1024, m0 + 16, lane),
                 laneA(Comb, 1024, m0 + 32, lane), laneA(Comb, 1024, m0 + 48, lane),
                 laneB(WoT, 1024, n0, lane),       laneB(WoT, 1024, n0 + 16, lane),
                 acc);
#pragma unroll
  for (int i = 0; i < 4; ++i)
#pragma unroll
    for (int j = 0; j < 2; ++j) {
      const int col = n0 + j * 16 + (lane & 15);
      const float bias = bo[col];
#pragma unroll
      for (int r = 0; r < 8; ++r) {
        const int m = m0 + i * 16 + (lane >> 4) * 8 + r;
        const size_t idx = (size_t)m * 1024 + col;
        R[idx] = acc[i][j][r] + bias + x[idx];
      }
    }
}

// One wave32 per row: mean/var over 1024, normalize, gamma/beta.
__global__ __launch_bounds__(256)
void k_layernorm(const float* __restrict__ R,
                 const float* __restrict__ gamma,
                 const float* __restrict__ beta,
                 float* __restrict__ out) {
  const int row  = blockIdx.x * 8 + (threadIdx.x >> 5);
  const int lane = threadIdx.x & 31;
  const float* r = R + (size_t)row * 1024;
  float vals[32];
  float s = 0.0f, ss = 0.0f;
#pragma unroll
  for (int t = 0; t < 32; ++t) {
    const float v = r[lane + 32 * t];
    vals[t] = v; s += v; ss = fmaf(v, v, ss);
  }
#pragma unroll
  for (int off = 16; off > 0; off >>= 1) {
    s  += __shfl_down(s,  off, 32);
    ss += __shfl_down(ss, off, 32);
  }
  s  = __shfl(s,  0, 32);
  ss = __shfl(ss, 0, 32);
  const float mu  = s * (1.0f / 1024.0f);
  const float var = ss * (1.0f / 1024.0f) - mu * mu;
  const float inv = rsqrtf(var + 1e-3f);
#pragma unroll
  for (int t = 0; t < 32; ++t) {
    const int i = lane + 32 * t;
    out[(size_t)row * 1024 + i] = gamma[i] * (vals[t] - mu) * inv + beta[i];
  }
}

// ---------------------------------------------------------------------------
extern "C" void kernel_launch(void* const* d_in, const int* in_sizes, int n_in,
                              void* d_out, int out_size, void* d_ws, size_t ws_size,
                              hipStream_t stream) {
  (void)in_sizes; (void)n_in; (void)out_size; (void)ws_size;
  const float* x    = (const float*)d_in[0];
  const float* Wq   = (const float*)d_in[1];
  const float* bq   = (const float*)d_in[2];
  const float* Wk   = (const float*)d_in[3];
  const float* bk   = (const float*)d_in[4];
  const float* Wv   = (const float*)d_in[5];
  const float* bv   = (const float*)d_in[6];
  const float* Wq2  = (const float*)d_in[7];
  const float* bq2  = (const float*)d_in[8];
  const float* Wk2  = (const float*)d_in[9];
  const float* bk2  = (const float*)d_in[10];
  const float* Wv2  = (const float*)d_in[11];
  const float* bv2  = (const float*)d_in[12];
  const float* Wo   = (const float*)d_in[13];
  const float* bo   = (const float*)d_in[14];
  const float* gmma = (const float*)d_in[15];
  const float* beta = (const float*)d_in[16];
  float* out = (float*)d_out;

  char* ws = (char*)d_ws;
  size_t off = 0;
  auto bump = [&](size_t bytes) -> void* {
    void* p = ws + off;
    off = (off + bytes + 255) & ~(size_t)255;
    return p;
  };
  unsigned short* xbf    = (unsigned short*)bump(16384ull * 1024 * 2);
  unsigned short* Wabf   = (unsigned short*)bump(3ull * 1024 * 1024 * 2);
  unsigned short* W2T    = (unsigned short*)bump(3ull * 1024 * 1024 * 2);
  unsigned short* WoT    = (unsigned short*)bump(1024ull * 1024 * 2);
  unsigned short* WfT    = (unsigned short*)bump(3ull * 1024 * 1024 * 2);
  float*          bfused = (float*)bump(3ull * 1024 * 4);
  unsigned short* Qt     = (unsigned short*)bump(64ull * 64 * 4096 * 2);
  unsigned short* Kt     = (unsigned short*)bump(64ull * 64 * 4096 * 2);
  unsigned short* Vh     = (unsigned short*)bump(64ull * 4096 * 64 * 2);
  unsigned short* St     = (unsigned short*)bump(64ull * 64 * 64 * 2);
  float*          qks    = (float*)bump(64ull * 4096 * 4);
  unsigned short* Comb   = (unsigned short*)bump(16384ull * 1024 * 2);
  float*          R      = (float*)bump(16384ull * 1024 * 4);

  // 1) fp32 -> bf16 conversions (x and A-side weights), transposed B-side weights
  k_cvt_bf16<<<dim3(65536), 256, 0, stream>>>(x, xbf, 16384 * 1024);
  k_cvt_bf16<<<dim3(4096), 256, 0, stream>>>(Wq, Wabf + 0ull * 1024 * 1024, 1024 * 1024);
  k_cvt_bf16<<<dim3(4096), 256, 0, stream>>>(Wk, Wabf + 1ull * 1024 * 1024, 1024 * 1024);
  k_cvt_bf16<<<dim3(4096), 256, 0, stream>>>(Wv, Wabf + 2ull * 1024 * 1024, 1024 * 1024);
  k_transpose_cvt<<<dim3(4096), 256, 0, stream>>>(Wq2, W2T + 0ull * 1024 * 1024);
  k_transpose_cvt<<<dim3(4096), 256, 0, stream>>>(Wk2, W2T + 1ull * 1024 * 1024);
  k_transpose_cvt<<<dim3(4096), 256, 0, stream>>>(Wv2, W2T + 2ull * 1024 * 1024);
  k_transpose_cvt<<<dim3(4096), 256, 0, stream>>>(Wo, WoT);

  // 2) fused biases: bfused = b1 @ W2 + b2  (exact, fp32)
  k_bias_fuse<<<dim3(4), 256, 0, stream>>>(bq, Wq2, bq2, bfused + 0);
  k_bias_fuse<<<dim3(4), 256, 0, stream>>>(bk, Wk2, bk2, bfused + 1024);
  k_bias_fuse<<<dim3(4), 256, 0, stream>>>(bv, Wv2, bv2, bfused + 2048);

  // 3) fused weights via WMMA (3 x 1024^3), stored [n][k] for B-operand use
  k_fuse_weights<<<dim3(8, 8, 3), 256, 0, stream>>>(Wabf, W2T, WfT);

  // 4) Q/K/V projection GEMM (16384 x 3072) with elu+1 + per-head scatter
  k_gemm_qkv<<<dim3(24, 128), 256, 0, stream>>>(xbf, WfT, bfused, Qt, Kt, Vh);

  // 5) per-head kernelized-attention state + row dots
  k_attn_state<<<dim3(64), 128, 0, stream>>>(Qt, Kt, St);
  k_qks<<<dim3(16, 64), 256, 0, stream>>>(Qt, Kt, qks);

  // 6) out = V @ S / qks, combined back to model layout
  k_attn_out<<<dim3(32, 64), 256, 0, stream>>>(Vh, St, qks, Comb);

  // 7) output projection + residual, then LayerNorm
  k_gemm_out<<<dim3(8, 128), 256, 0, stream>>>(Comb, WoT, bo, x, R);
  k_layernorm<<<dim3(2048), 256, 0, stream>>>(R, gmma, beta, out);
}
